// GInvariantLSTMLayer_59399397704121
// MI455X (gfx1250) — matrix operand order repaired and checked by reading, not compiled, not run on hardware
//
#include <hip/hip_runtime.h>
#include <stddef.h>
#include <stdint.h>

typedef __attribute__((ext_vector_type(16))) __bf16 v16bf;
typedef __attribute__((ext_vector_type(8)))  float  v8f;
typedef int v4i __attribute__((vector_size(16)));   // matches async-LDS builtin param

#define H_DIM    2048
#define L_SEQ    512
#define NBLOCKS  64
#define SC_BYTES 65536           // staged B bytes per super-chunk per block (8 kc)
#define NSC      8               // super-chunks per step (8 kc * K32 * 8sc = K2048)

// ---------------- workspace layout (bytes) ----------------
// s     : [0, 131072)            float  s[t*64+m]
// bar   : [131072, 131328)       grid barrier counter
// hbuf  : [131328, 655616)       __bf16 hbuf[2][64][2048]
// Wpk   : [655616, 34210048)     __bf16 packed WMMA B fragments (per-block contiguous)
#define OFF_S    0
#define OFF_BAR  131072
#define OFF_HBUF 131328
#define OFF_WPK  655616

#if defined(__has_builtin)
#  if __has_builtin(__builtin_amdgcn_global_load_async_to_lds_b128)
#    define HAVE_ASYNC_LDS 1
#  endif
#  if __has_builtin(__builtin_amdgcn_s_wait_asynccnt)
#    define HAVE_WAIT_ASYNC 1
#  endif
#endif

__device__ __forceinline__ float sigmoidf_(float x) {
    return 1.0f / (1.0f + __expf(-x));
}
__device__ __forceinline__ float tanhf_(float x) {
    x = fminf(15.0f, fmaxf(-15.0f, x));
    float e = __expf(2.0f * x);
    return (e - 1.0f) / (e + 1.0f);
}

// issue one 16-byte global -> LDS copy (CDNA5 async path when available)
__device__ __forceinline__ void copy16_to_lds(const char* gsrc, char* lds) {
#if defined(HAVE_ASYNC_LDS)
    __builtin_amdgcn_global_load_async_to_lds_b128(
        (__attribute__((address_space(1))) v4i*)(void*)gsrc,
        (__attribute__((address_space(3))) v4i*)(void*)lds, 0, 0);
#else
    *(uint4*)lds = *(const uint4*)gsrc;
#endif
}

__device__ __forceinline__ void wait_async_prev16() {
#if defined(HAVE_ASYNC_LDS)
#  if defined(HAVE_WAIT_ASYNC)
    __builtin_amdgcn_s_wait_asynccnt(16);
#  else
    asm volatile("s_wait_asynccnt 0x10" ::: "memory");
#  endif
#endif
}
__device__ __forceinline__ void wait_async_all() {
#if defined(HAVE_ASYNC_LDS)
#  if defined(HAVE_WAIT_ASYNC)
    __builtin_amdgcn_s_wait_asynccnt(0);
#  else
    asm volatile("s_wait_asynccnt 0x0" ::: "memory");
#  endif
#endif
}

// ---- prepass 1: s[t*64+m] = sum_d X[m][t][d]; one wave per (m,t) row ----
__global__ void ginv_prep_sums(const float* __restrict__ X, float* __restrict__ s) {
    const int row  = blockIdx.x * 8 + (threadIdx.x >> 5);   // m*512 + t
    const int lane = threadIdx.x & 31;
    const float* xr = X + (size_t)row * 1024;
    float acc = 0.0f;
#pragma unroll
    for (int j = 0; j < 32; ++j) acc += xr[lane + 32 * j];
#pragma unroll
    for (int off = 16; off > 0; off >>= 1) acc += __shfl_down(acc, off, 32);
    if (lane == 0) {
        const int m = row >> 9;
        const int t = row & 511;
        s[t * 64 + m] = acc;
    }
}

// ---- prepass 2: pack lin_W fp32 [4][k_out][h_in] into per-lane bf16 B fragments.
// Fragment (K=32 bf16 B): lane<16 -> N=lane, K=0..15; lane>=16 -> N=lane-16, K=16..31.
// Layout: [blk(64)][sc(8)][ctlo(2)][g(4)][j(8)][lane(32)][e(16)]  (blk region contiguous)
__global__ void ginv_prep_pack(const float* __restrict__ W, __bf16* __restrict__ Wpk) {
    const int tid  = blockIdx.x * 256 + threadIdx.x;   // 2^20 threads
    const int lane = tid & 31;
    const int kc   = (tid >> 5) & 63;
    const int ct   = (tid >> 11) & 127;
    const int g    = tid >> 18;
    const int half = lane >> 4, ncol = lane & 15;
    const int nOut = ct * 16 + ncol;
    const int k    = kc * 32 + half * 16;
    const int blk  = ct >> 1, ctlo = ct & 1;
    const int sc   = kc >> 3, j = kc & 7;
    const float* src = W + ((size_t)g * H_DIM + nOut) * H_DIM + k;
    __bf16* dst = Wpk +
        ((((((size_t)blk * 8 + sc) * 2 + ctlo) * 4 + g) * 8 + j) * 32 + lane) * 16;
#pragma unroll
    for (int e = 0; e < 16; ++e) dst[e] = (__bf16)src[e];
}

// ---- prepass 3: h0 -> bf16 buffer 0, reset barrier ----
__global__ void ginv_prep_state(const float* __restrict__ H0, __bf16* __restrict__ hbuf,
                                unsigned int* __restrict__ bar) {
    const int i = blockIdx.x * 256 + threadIdx.x;      // 131072 threads
    hbuf[i] = (__bf16)H0[i];
    if (i == 0) *bar = 0u;
}

// ---- persistent LSTM kernel: 64 blocks x 256 threads, 192KB LDS (3 B-buffers) ----
__global__ __launch_bounds__(256, 1)
void ginv_lstm_main(const float* __restrict__ s,       // [512][64]
                    const __bf16* __restrict__ Wpk,
                    const float* __restrict__ inv_w,   // [4][2048]
                    const float* __restrict__ inv_b,   // [4][2048]
                    const float* __restrict__ lin_b,   // [4][2048]
                    __bf16* __restrict__ hbuf,         // [2][64][2048]
                    float* __restrict__ out,           // [2048][64]
                    unsigned int* __restrict__ bar) {
    extern __shared__ char smem[];                     // 3 * SC_BYTES
    const int tid   = threadIdx.x;
    const int lane  = tid & 31;
    const int wv    = tid >> 5;
    const int Mtile = wv >> 1;                 // 0..3 (16 batch rows each)
    const int ctlo  = wv & 1;
    const int ctIdx = blockIdx.x * 2 + ctlo;   // 0..127
    const int ncol  = lane & 15;
    const int half  = lane >> 4;
    const int n     = ctIdx * 16 + ncol;       // hidden output index (fixed per lane)

    // t-invariant per-lane gate params
    float wI[4], bI[4], bL[4];
#pragma unroll
    for (int g = 0; g < 4; ++g) {
        wI[g] = inv_w[g * H_DIM + n];
        bI[g] = inv_b[g * H_DIM + n];
        bL[g] = lin_b[g * H_DIM + n];
    }

    // cell state lives in registers for the whole sequence
    float cst[8];
#pragma unroll
    for (int r = 0; r < 8; ++r) cst[r] = 0.0f;

    const char* WpkB = (const char*)Wpk + (size_t)blockIdx.x * (NSC * SC_BYTES);

    union Frag { v16bf v; uint4 q[2]; };

    // prologue: stage super-chunk 0 into LDS buffer 0 (16 x 16B per thread)
#pragma unroll
    for (int it = 0; it < 16; ++it)
        copy16_to_lds(WpkB + (size_t)(it * 256 + tid) * 16,
                      smem + (size_t)(it * 256 + tid) * 16);

    int bufCur = 0;
    int scg = 0;
    for (int t = 0; t < L_SEQ; ++t) {
        const __bf16* hr = hbuf + (size_t)(t & 1) * (64 * H_DIM);
        __bf16* hw = hbuf + (size_t)((t + 1) & 1) * (64 * H_DIM);
        const __bf16* arow = hr + (size_t)(Mtile * 16 + ncol) * H_DIM + half * 16;

        v8f acc[4] = {};

        for (int sc = 0; sc < NSC; ++sc, ++scg) {
            // issue async staging for the NEXT super-chunk into the third buffer
            const int bufNext = (bufCur == 2) ? 0 : bufCur + 1;
            if (scg + 1 < L_SEQ * NSC) {
                const char* gs = WpkB + (size_t)((scg + 1) & (NSC - 1)) * SC_BYTES;
                char* ld = smem + (size_t)bufNext * SC_BYTES;
#pragma unroll
                for (int it = 0; it < 16; ++it)
                    copy16_to_lds(gs + (size_t)(it * 256 + tid) * 16,
                                  ld + (size_t)(it * 256 + tid) * 16);
                wait_async_prev16();   // in-order: current super-chunk landed
            } else {
                wait_async_all();
            }
            __syncthreads();

            const char* bbase = smem + (size_t)bufCur * SC_BYTES +
                                (size_t)ctlo * (4 * 8 * 1024) + lane * 32;
#pragma unroll
            for (int j = 0; j < 8; ++j) {
                const int k0 = (sc * 8 + j) * 32;
                Frag a;
                a.q[0] = *(const uint4*)(arow + k0);
                a.q[1] = *(const uint4*)(arow + k0 + 8);
                // load all 4 gate B-fragments into distinct registers first so the
                // 4 WMMAs can issue back-to-back behind a single s_wait_dscnt
                const char* bp = bbase + (size_t)j * 1024;
                Frag b0, b1, b2, b3;
                b0.q[0] = *(const uint4*)(bp);
                b0.q[1] = *(const uint4*)(bp + 16);
                b1.q[0] = *(const uint4*)(bp + 8192);
                b1.q[1] = *(const uint4*)(bp + 8208);
                b2.q[0] = *(const uint4*)(bp + 16384);
                b2.q[1] = *(const uint4*)(bp + 16400);
                b3.q[0] = *(const uint4*)(bp + 24576);
                b3.q[1] = *(const uint4*)(bp + 24592);
                acc[0] = __builtin_amdgcn_wmma_f32_16x16x32_bf16(
                    false, a.v, false, b0.v, (short)0, acc[0], false, false);
                acc[1] = __builtin_amdgcn_wmma_f32_16x16x32_bf16(
                    false, a.v, false, b1.v, (short)0, acc[1], false, false);
                acc[2] = __builtin_amdgcn_wmma_f32_16x16x32_bf16(
                    false, a.v, false, b2.v, (short)0, acc[2], false, false);
                acc[3] = __builtin_amdgcn_wmma_f32_16x16x32_bf16(
                    false, a.v, false, b3.v, (short)0, acc[3], false, false);
            }
            bufCur = bufNext;
        }

        // elementwise gate update; D element r of v8f is M = half*8 + r, N = ncol
        const float* st = s + t * 64;
#pragma unroll
        for (int r = 0; r < 8; ++r) {
            const int m = Mtile * 16 + half * 8 + r;
            const float sv = st[m];
            float z0 = acc[0][r] + bL[0] + sv * wI[0] + bI[0];
            float z1 = acc[1][r] + bL[1] + sv * wI[1] + bI[1];
            float z2 = acc[2][r] + bL[2] + sv * wI[2] + bI[2];
            float z3 = acc[3][r] + bL[3] + sv * wI[3] + bI[3];
            const float ig = sigmoidf_(z0);
            const float fg = sigmoidf_(z1);
            const float gg = tanhf_(z2);
            const float og = sigmoidf_(z3);
            const float c = fg * cst[r] + ig * gg;
            cst[r] = c;
            const float h = og * tanhf_(c);
            hw[(size_t)m * H_DIM + n] = (__bf16)h;
            if (t == L_SEQ - 1) out[(size_t)n * 64 + m] = h;   // output is h.T
        }

        // device-wide barrier (monotonic counter), release h writes first
        __threadfence();
        __syncthreads();
        if (tid == 0) {
            unsigned int g = __atomic_fetch_add(bar, 1u, __ATOMIC_ACQ_REL);
            unsigned int target = (g / NBLOCKS + 1u) * NBLOCKS;
            while (__atomic_load_n(bar, __ATOMIC_ACQUIRE) < target)
                __builtin_amdgcn_s_sleep(8);
        }
        __syncthreads();
        __builtin_amdgcn_fence(__ATOMIC_ACQUIRE, "agent");
    }
}

extern "C" void kernel_launch(void* const* d_in, const int* in_sizes, int n_in,
                              void* d_out, int out_size, void* d_ws, size_t ws_size,
                              hipStream_t stream) {
    const float* X     = (const float*)d_in[0];
    const float* H0    = (const float*)d_in[1];
    const float* inv_w = (const float*)d_in[2];
    const float* inv_b = (const float*)d_in[3];
    const float* lin_W = (const float*)d_in[4];
    const float* lin_b = (const float*)d_in[5];
    float* out = (float*)d_out;

    char* ws = (char*)d_ws;
    float*        s    = (float*)(ws + OFF_S);
    unsigned int* bar  = (unsigned int*)(ws + OFF_BAR);
    __bf16*       hbuf = (__bf16*)(ws + OFF_HBUF);
    __bf16*       Wpk  = (__bf16*)(ws + OFF_WPK);

    ginv_prep_sums<<<4096, 256, 0, stream>>>(X, s);
    ginv_prep_pack<<<4096, 256, 0, stream>>>(lin_W, Wpk);
    ginv_prep_state<<<512, 256, 0, stream>>>(H0, hbuf, bar);
    ginv_lstm_main<<<NBLOCKS, 256, 3 * SC_BYTES, stream>>>(s, Wpk, inv_w, inv_b,
                                                           lin_b, hbuf, out, bar);
}
